// SkewSymmetricMatrix_54039278518768
// MI455X (gfx1250) — compile-verified, matrix operand order checked by reading, not run on hardware
//
#include <hip/hip_runtime.h>
#include <hip/hip_bf16.h>
#include <math.h>

typedef __attribute__((ext_vector_type(16))) _Float16 v16h;
typedef __attribute__((ext_vector_type(8)))  float    v8f;

union V16 { v16h v; uint4 u[2]; };

#define BATCH 16384
#define DIM   64
#define HID   128
#define NOUT  4096          // 64*64, flattened J row
#define WPB   4             // waves per block
#define RPW   16            // batch rows per wave (WMMA M)
#define NHALF 2048          // N columns per wave (half of NOUT)
#define NTILES (NHALF / 16) // 128 n-tiles per wave

// Hardware tanh (V_TANH_F32 on gfx1250) with safe fallback.
static __device__ inline float fast_tanh(float v) {
#if __has_builtin(__builtin_amdgcn_tanhf)
    return __builtin_amdgcn_tanhf(v);
#elif __has_builtin(__builtin_amdgcn_tanh_f32)
    return __builtin_amdgcn_tanh_f32(v);
#else
    return tanhf(v);
#endif
}

// p(i,j) for i<j in np.triu_indices(64, k=1) row-major order
__device__ __host__ inline int triu_idx(int i, int j) {
    return i * 63 - (i * (i - 1)) / 2 + (j - i - 1);
}

// Build folded weights: W2J[4096][128] f16, W1h[128][64] f16, b2J[4096] f32
__global__ void prep_kernel(const float* __restrict__ W1,
                            const float* __restrict__ W2,
                            const float* __restrict__ b2,
                            _Float16* __restrict__ W2J,
                            _Float16* __restrict__ W1h,
                            float* __restrict__ b2J) {
    int t = blockIdx.x * blockDim.x + threadIdx.x;
    if (t < NOUT * HID) {
        int q = t >> 7;          // flattened (i,j)
        int k = t & 127;
        int i = q >> 6, j = q & 63;
        float v = 0.0f;
        if (i < j)      v =  W2[(size_t)triu_idx(i, j) * HID + k];
        else if (i > j) v = -W2[(size_t)triu_idx(j, i) * HID + k];
        W2J[t] = (_Float16)v;
    }
    if (t < HID * DIM) W1h[t] = (_Float16)W1[t];
    if (t < NOUT) {
        int i = t >> 6, j = t & 63;
        float v = 0.0f;
        if (i < j)      v =  b2[triu_idx(i, j)];
        else if (i > j) v = -b2[triu_idx(j, i)];
        b2J[t] = v;
    }
}

__device__ inline v16h cvt16(float4 p0, float4 p1, float4 p2, float4 p3) {
    v16h a;
    a[0] = (_Float16)p0.x; a[1] = (_Float16)p0.y; a[2]  = (_Float16)p0.z; a[3]  = (_Float16)p0.w;
    a[4] = (_Float16)p1.x; a[5] = (_Float16)p1.y; a[6]  = (_Float16)p1.z; a[7]  = (_Float16)p1.w;
    a[8] = (_Float16)p2.x; a[9] = (_Float16)p2.y; a[10] = (_Float16)p2.z; a[11] = (_Float16)p2.w;
    a[12] = (_Float16)p3.x; a[13] = (_Float16)p3.y; a[14] = (_Float16)p3.z; a[15] = (_Float16)p3.w;
    return a;
}

__global__ __launch_bounds__(WPB * 32) void jacgemm_kernel(
    const float* __restrict__ x,      // [16384,64]
    const float* __restrict__ b1,     // [128]
    const _Float16* __restrict__ W1h, // [128,64]
    const _Float16* __restrict__ W2J, // [4096,128]
    const float* __restrict__ b2J,    // [4096]
    float* __restrict__ out)          // [16384,4096]
{
    __shared__ _Float16 hstage[WPB][RPW * HID];   // 4 KB per wave (private copy)

    const int lane  = threadIdx.x & 31;
    const int wv    = threadIdx.x >> 5;
    const int b0    = blockIdx.x * (2 * RPW) + (wv >> 1) * RPW;  // 2 batch tiles/block
    const int nbase = (wv & 1) * NHALF;                          // N half per wave
    const int m     = lane & 15;      // row (A) / column (B) within tile
    const int hi    = lane >> 4;      // 0 or 1: which K half this lane holds
    const int koff  = hi << 3;        // lane K base offset within fragment

    // ---------------- GEMM1: h = tanh(x @ W1^T + b1) ----------------
    // (recomputed redundantly by both waves of a batch-tile pair; 16 WMMAs, cheap)
    V16 a1[2];
    {
        const float* xr = x + (size_t)(b0 + m) * DIM;
        #pragma unroll
        for (int f = 0; f < 2; ++f) {
            int k0 = f * 32 + koff;
            float4 p0 = *(const float4*)(xr + k0);
            float4 p1 = *(const float4*)(xr + k0 + 4);
            float4 p2 = *(const float4*)(xr + k0 + 16);
            float4 p3 = *(const float4*)(xr + k0 + 20);
            a1[f].v = cvt16(p0, p1, p2, p3);
        }
    }
    #pragma unroll
    for (int nt = 0; nt < HID / 16; ++nt) {
        int n = nt * 16 + m;
        const _Float16* wr = W1h + (size_t)n * DIM;
        v8f c = {0.f, 0.f, 0.f, 0.f, 0.f, 0.f, 0.f, 0.f};
        #pragma unroll
        for (int f = 0; f < 2; ++f) {
            V16 bf;
            int k0 = f * 32 + koff;
            bf.u[0] = *(const uint4*)(wr + k0);
            bf.u[1] = *(const uint4*)(wr + k0 + 16);
            c = __builtin_amdgcn_wmma_f32_16x16x32_f16(
                    false, a1[f].v, false, bf.v, (short)0, c, false, false);
        }
        float bv = b1[n];
        #pragma unroll
        for (int r = 0; r < 8; ++r) {
            int mm = r + 8 * hi;                  // C layout: lanes 16-31 hold M=r+8
            hstage[wv][mm * HID + n] = (_Float16)fast_tanh(c[r] + bv);
        }
    }
    __syncthreads();

    // Reload h in A-fragment layout from LDS (ds_load_b128), K = 128 -> 4 frags
    V16 a2[4];
    {
        const _Float16* hp = &hstage[wv][m * HID];
        #pragma unroll
        for (int f = 0; f < 4; ++f) {
            int k0 = f * 32 + koff;
            a2[f].u[0] = *(const uint4*)(hp + k0);
            a2[f].u[1] = *(const uint4*)(hp + k0 + 16);
        }
    }

    // ---------------- GEMM2: J = h @ W2J^T + b2J (coalesced stores) --------
    // Double-buffered B fragments: tile nt+1's loads are in flight while tile
    // nt's WMMAs and stores execute.
    const _Float16* wbase = W2J + (size_t)(nbase + m) * HID;  // row of tile 0
    const float*    bbase = b2J + nbase + m;
    float* orow0 = out + (size_t)(b0 + 8 * hi) * NOUT + nbase + m;

    V16 cur[4], nxt[4];
    float bcur, bnxt;
    #pragma unroll
    for (int f = 0; f < 4; ++f) {
        int k0 = f * 32 + koff;
        cur[f].u[0] = *(const uint4*)(wbase + k0);
        cur[f].u[1] = *(const uint4*)(wbase + k0 + 16);
    }
    bcur = bbase[0];

    #pragma unroll 2
    for (int nt = 0; nt < NTILES; ++nt) {
        if (nt + 1 < NTILES) {                       // prefetch next tile's B + bias
            const _Float16* wr = wbase + (size_t)(nt + 1) * 16 * HID;
            #pragma unroll
            for (int f = 0; f < 4; ++f) {
                int k0 = f * 32 + koff;
                nxt[f].u[0] = *(const uint4*)(wr + k0);
                nxt[f].u[1] = *(const uint4*)(wr + k0 + 16);
            }
            bnxt = bbase[(nt + 1) * 16];
        }
        v8f c = {0.f, 0.f, 0.f, 0.f, 0.f, 0.f, 0.f, 0.f};
        #pragma unroll
        for (int f = 0; f < 4; ++f)
            c = __builtin_amdgcn_wmma_f32_16x16x32_f16(
                    false, a2[f].v, false, cur[f].v, (short)0, c, false, false);
        #pragma unroll
        for (int r = 0; r < 8; ++r)
            orow0[(size_t)r * NOUT + nt * 16] = c[r] + bcur;  // 16 lanes -> contiguous 64B
        #pragma unroll
        for (int f = 0; f < 4; ++f) cur[f] = nxt[f];
        bcur = bnxt;
    }
}

extern "C" void kernel_launch(void* const* d_in, const int* in_sizes, int n_in,
                              void* d_out, int out_size, void* d_ws, size_t ws_size,
                              hipStream_t stream) {
    const float* x  = (const float*)d_in[0];
    const float* W1 = (const float*)d_in[1];
    const float* b1 = (const float*)d_in[2];
    const float* W2 = (const float*)d_in[3];
    const float* b2 = (const float*)d_in[4];
    float* out = (float*)d_out;

    char* ws = (char*)d_ws;
    _Float16* W2J = (_Float16*)ws;                         // 4096*128*2 = 1 MiB
    _Float16* W1h = (_Float16*)(ws + 1048576);             // 128*64*2   = 16 KiB
    float*    b2J = (float*)   (ws + 1048576 + 16384);     // 4096*4     = 16 KiB

    int prep_threads = NOUT * HID;                         // 524288
    prep_kernel<<<(prep_threads + 255) / 256, 256, 0, stream>>>(W1, W2, b2, W2J, W1h, b2J);

    int blocks = BATCH / (2 * RPW);                        // 512 blocks, 4 waves each
    jacgemm_kernel<<<blocks, WPB * 32, 0, stream>>>(x, b1, W1h, W2J, b2J, out);
}
